// Video_multilevel_encoding_2516850835457
// MI455X (gfx1250) — matrix operand order repaired and checked by reading, not compiled, not run on hardware
//
#include <hip/hip_runtime.h>
#include <hip/hip_bf16.h>

// ---------------------------------------------------------------------------
// CDNA5 (gfx1250, wave32). All GEMM work: v_wmma_f32_16x16x32_bf16.
// Block tile 64(M) x 128(N): 8 waves, each wave = 4 M-tiles x 1 N-tile.
// A staged via global_load_async_to_lds_b128 (ASYNCcnt path) when available,
// B staged transposed so each fragment is two 16B LDS reads (k=8h, k=16+8h).
// ---------------------------------------------------------------------------

typedef __attribute__((ext_vector_type(16))) __bf16 v16bf;
typedef __attribute__((ext_vector_type(8)))  float  v8f;

#if defined(__gfx1250__) && __has_builtin(__builtin_amdgcn_global_load_async_to_lds_b128) && __has_builtin(__builtin_amdgcn_s_wait_asynccnt)
#define USE_ASYNC_LDS 1
typedef int async_i4 __attribute__((vector_size(16)));   // matches builtin param type
#else
#define USE_ASYNC_LDS 0
#endif

typedef union { v16bf v; uint4 q[2]; } frag_u;

// ---------------------------------------------------------------------------
// C[z] = A[z] (MxK, bf16, row-major) * B[z] (KxN, bf16, row-major) + bias
// grid = (N/128, M/64, Z), block = 256
// ---------------------------------------------------------------------------
__global__ __launch_bounds__(256)
void gemm_wmma_bf16(const __bf16* __restrict__ A, const __bf16* __restrict__ Bm,
                    const float* __restrict__ bias, float* __restrict__ C,
                    int M, int N, int Kd,
                    long long sA, long long sB, long long sBias, long long sC) {
    __shared__ __attribute__((aligned(32))) __bf16 aT[64 * 32];   // [row][k]
    __shared__ __attribute__((aligned(32))) __bf16 bT[128 * 32];  // [col][k] (transposed)

    const int z = blockIdx.z;
    A  += (size_t)z * sA;
    Bm += (size_t)z * sB;
    C  += (size_t)z * sC;

    const int tid  = threadIdx.x;
    const int lane = tid & 31;
    const int wave = tid >> 5;
    const int l15  = lane & 15;
    const int half = lane >> 4;              // 0: lanes 0-15, 1: lanes 16-31
    const int row0 = blockIdx.y * 64;
    const int colB = blockIdx.x * 128;       // block N base
    const int col0 = colB + wave * 16;       // wave N base

    // ---- staging pointers (advance by 32 / 32*N per K-step) ----------------
    const int ar = tid >> 2, aq = tid & 3;               // A: row, 8-elem quad
    const __bf16* pA  = A + (size_t)(row0 + ar) * Kd + aq * 8;
    __bf16* aDst      = &aT[ar * 32 + aq * 8];

    const int k0r = tid >> 4, cq0 = tid & 15;            // B it0: k-row, col quad
    const int k1r = (tid + 256) >> 4, cq1 = (tid + 256) & 15;
    const __bf16* pB0 = Bm + (size_t)k0r * N + colB + cq0 * 8;
    const __bf16* pB1 = Bm + (size_t)k1r * N + colB + cq1 * 8;

    // ---- fragment pointers (fixed) -----------------------------------------
    const __bf16* fbSrc = &bT[(wave * 16 + l15) * 32 + half * 8]; // 8 bf16 at k=8h, 8 at k=16+8h
    const int kb = half * 8;
    const __bf16* faSrc[4];
    #pragma unroll
    for (int mt = 0; mt < 4; ++mt) faSrc[mt] = &aT[(mt * 16 + l15) * 32 + kb];

    v8f acc0 = {}, acc1 = {}, acc2 = {}, acc3 = {};

    for (int k0 = 0; k0 < Kd; k0 += 32) {
        // A tile: 64x32 bf16 = 4KB -> one b128 per thread
#if USE_ASYNC_LDS
        __builtin_amdgcn_global_load_async_to_lds_b128(
            (async_i4*)const_cast<__bf16*>(pA), (async_i4*)aDst, 0, 0);
#else
        *(uint4*)aDst = *(const uint4*)pA;
#endif
        // B tile: 32x128 bf16 = 8KB -> two b128 per thread, stored transposed
        {
            uint4 v0 = *(const uint4*)pB0;
            uint4 v1 = *(const uint4*)pB1;
            const __bf16* e0 = (const __bf16*)&v0;
            const __bf16* e1 = (const __bf16*)&v1;
            #pragma unroll
            for (int j = 0; j < 8; ++j) bT[(cq0 * 8 + j) * 32 + k0r] = e0[j];
            #pragma unroll
            for (int j = 0; j < 8; ++j) bT[(cq1 * 8 + j) * 32 + k1r] = e1[j];
        }
        if (k0 + 32 < Kd) __builtin_prefetch(pB0 + (size_t)32 * N, 0, 0);
        pA += 32;
        pB0 += (size_t)32 * N;
        pB1 += (size_t)32 * N;
#if USE_ASYNC_LDS
        __builtin_amdgcn_s_wait_asynccnt(0);
#endif
        __syncthreads();

        frag_u fbu; fbu.q[0] = *(const uint4*)(fbSrc); fbu.q[1] = *(const uint4*)(fbSrc + 16);
        const v16bf fb = fbu.v;
        frag_u a0, a1, a2, a3;
        a0.q[0] = *(const uint4*)(faSrc[0]);      a0.q[1] = *(const uint4*)(faSrc[0] + 16);
        a1.q[0] = *(const uint4*)(faSrc[1]);      a1.q[1] = *(const uint4*)(faSrc[1] + 16);
        a2.q[0] = *(const uint4*)(faSrc[2]);      a2.q[1] = *(const uint4*)(faSrc[2] + 16);
        a3.q[0] = *(const uint4*)(faSrc[3]);      a3.q[1] = *(const uint4*)(faSrc[3] + 16);

        acc0 = __builtin_amdgcn_wmma_f32_16x16x32_bf16(false, a0.v, false, fb, (short)0, acc0, false, false);
        acc1 = __builtin_amdgcn_wmma_f32_16x16x32_bf16(false, a1.v, false, fb, (short)0, acc1, false, false);
        acc2 = __builtin_amdgcn_wmma_f32_16x16x32_bf16(false, a2.v, false, fb, (short)0, acc2, false, false);
        acc3 = __builtin_amdgcn_wmma_f32_16x16x32_bf16(false, a3.v, false, fb, (short)0, acc3, false, false);
        __syncthreads();
    }

    const float bv = bias ? (bias + (size_t)z * sBias)[col0 + l15] : 0.0f;
    v8f accs[4] = {acc0, acc1, acc2, acc3};
    #pragma unroll
    for (int mt = 0; mt < 4; ++mt) {
        float* Cp = C + (size_t)(row0 + mt * 16 + half * 8) * N + col0 + l15;
        #pragma unroll
        for (int r = 0; r < 8; ++r) Cp[(size_t)r * N] = accs[mt][r] + bv;
    }
}

// ---------------------------------------------------------------------------
// Temporal conv as shifted-A WMMA GEMM fused with ReLU + time-max-pool via
// f32-bit atomicMax (ReLU >= 0 so int-bit compare is order preserving).
// X: [B,T,2H] bf16; CW: [w*2H, K] bf16; conout: [B, 4*K] f32 (zero-init).
// grid = (K/128, ceil((T+w-1)/64), B), block = 256
// ---------------------------------------------------------------------------
__global__ __launch_bounds__(256)
void conv_wmma(const __bf16* __restrict__ X, const __bf16* __restrict__ CW,
               const float* __restrict__ cbias, float* __restrict__ conout,
               int w, int koff) {
    __shared__ __attribute__((aligned(32))) __bf16 aT[64 * 32];
    __shared__ __attribute__((aligned(32))) __bf16 bT[128 * 32];
    const int Tt = 128, HH2 = 1024, Kc = 512;

    const int b    = blockIdx.z;
    const int tid  = threadIdx.x;
    const int lane = tid & 31;
    const int wave = tid >> 5;
    const int l15  = lane & 15;
    const int half = lane >> 4;
    const int row0 = blockIdx.y * 64;                 // t' base
    const int colB = blockIdx.x * 128;
    const int col0 = colB + wave * 16;
    const int Tout = Tt + w - 1;
    const int Kd   = w * HH2;

    const int ar = tid >> 2, aq = tid & 3;
    const int k0r = tid >> 4, cq0 = tid & 15;
    const int k1r = (tid + 256) >> 4, cq1 = (tid + 256) & 15;
    const __bf16* pB0 = CW + (size_t)k0r * Kc + colB + cq0 * 8;
    const __bf16* pB1 = CW + (size_t)k1r * Kc + colB + cq1 * 8;

    const __bf16* fbSrc = &bT[(wave * 16 + l15) * 32 + half * 8];
    const int kb = half * 8;
    const __bf16* faSrc[4];
    #pragma unroll
    for (int mt = 0; mt < 4; ++mt) faSrc[mt] = &aT[(mt * 16 + l15) * 32 + kb];

    v8f acc0 = {}, acc1 = {}, acc2 = {}, acc3 = {};

    for (int k0 = 0; k0 < Kd; k0 += 32) {
        const int kh = k0 >> 10;                       // k0 / 2H
        const int d0 = k0 & (HH2 - 1);
        {   // A tile with time shift + zero padding (vectorized 16B per thread)
            const int t = row0 + ar - (w - 1) + kh;    // input time index
            uint4 v = {0u, 0u, 0u, 0u};
            if (t >= 0 && t < Tt)
                v = *(const uint4*)&X[((size_t)b * Tt + t) * HH2 + d0 + aq * 8];
            *(uint4*)&aT[ar * 32 + aq * 8] = v;
        }
        {   // B tile transposed
            uint4 v0 = *(const uint4*)pB0;
            uint4 v1 = *(const uint4*)pB1;
            const __bf16* e0 = (const __bf16*)&v0;
            const __bf16* e1 = (const __bf16*)&v1;
            #pragma unroll
            for (int j = 0; j < 8; ++j) bT[(cq0 * 8 + j) * 32 + k0r] = e0[j];
            #pragma unroll
            for (int j = 0; j < 8; ++j) bT[(cq1 * 8 + j) * 32 + k1r] = e1[j];
        }
        pB0 += (size_t)32 * Kc;
        pB1 += (size_t)32 * Kc;
        __syncthreads();

        frag_u fbu; fbu.q[0] = *(const uint4*)(fbSrc); fbu.q[1] = *(const uint4*)(fbSrc + 16);
        const v16bf fb = fbu.v;
        frag_u a0, a1, a2, a3;
        a0.q[0] = *(const uint4*)(faSrc[0]);      a0.q[1] = *(const uint4*)(faSrc[0] + 16);
        a1.q[0] = *(const uint4*)(faSrc[1]);      a1.q[1] = *(const uint4*)(faSrc[1] + 16);
        a2.q[0] = *(const uint4*)(faSrc[2]);      a2.q[1] = *(const uint4*)(faSrc[2] + 16);
        a3.q[0] = *(const uint4*)(faSrc[3]);      a3.q[1] = *(const uint4*)(faSrc[3] + 16);

        acc0 = __builtin_amdgcn_wmma_f32_16x16x32_bf16(false, a0.v, false, fb, (short)0, acc0, false, false);
        acc1 = __builtin_amdgcn_wmma_f32_16x16x32_bf16(false, a1.v, false, fb, (short)0, acc1, false, false);
        acc2 = __builtin_amdgcn_wmma_f32_16x16x32_bf16(false, a2.v, false, fb, (short)0, acc2, false, false);
        acc3 = __builtin_amdgcn_wmma_f32_16x16x32_bf16(false, a3.v, false, fb, (short)0, acc3, false, false);
        __syncthreads();
    }

    const float bias = cbias[col0 + l15];
    int* dst = (int*)conout + (size_t)b * 2048 + koff + col0 + l15;
    v8f accs[4] = {acc0, acc1, acc2, acc3};
    #pragma unroll
    for (int mt = 0; mt < 4; ++mt) {
        #pragma unroll
        for (int r = 0; r < 8; ++r) {
            const int tp = row0 + mt * 16 + half * 8 + r;
            if (tp < Tout) {
                float v = accs[mt][r] + bias;
                v = v > 0.0f ? v : 0.0f;
                atomicMax(dst, __float_as_int(v));
            }
        }
    }
}

// ---------------------------------------------------------------------------
// GRU gates (both directions fused)
// ---------------------------------------------------------------------------
__global__ __launch_bounds__(256)
void gru_gate(const float* __restrict__ xw_f, const float* __restrict__ xw_b,
              const float* __restrict__ gh,
              const float* __restrict__ bhh_f, const float* __restrict__ bhh_b,
              float* __restrict__ h, __bf16* __restrict__ hbf,
              float* __restrict__ hs_f, float* __restrict__ hs_b, int t) {
    const int Bn = 128, Tt = 128, Hh = 512;
    const int idx = blockIdx.x * 256 + threadIdx.x;   // 2*B*H
    const int dir = idx >> 16;
    const int r   = idx & 65535;
    const int b   = r >> 9;
    const int j   = r & 511;
    const int tt  = dir ? (Tt - 1 - t) : t;

    const float* xw  = dir ? xw_b : xw_f;
    const float* bhh = dir ? bhh_b : bhh_f;

    const size_t xbase = ((size_t)b * Tt + tt) * 1536;
    const float xr = xw[xbase + j];
    const float xz = xw[xbase + 512 + j];
    const float xn = xw[xbase + 1024 + j];

    const size_t gbase = ((size_t)dir * Bn + b) * 1536;
    const float hr = gh[gbase + j]        + bhh[j];
    const float hz = gh[gbase + 512 + j]  + bhh[512 + j];
    const float hn = gh[gbase + 1024 + j] + bhh[1024 + j];

    const float rg = 1.0f / (1.0f + __expf(-(xr + hr)));
    const float zg = 1.0f / (1.0f + __expf(-(xz + hz)));
    const float ng = tanhf(xn + rg * hn);

    const size_t hidx = ((size_t)dir * Bn + b) * Hh + j;
    const float hnew = (1.0f - zg) * ng + zg * h[hidx];
    h[hidx]   = hnew;
    hbf[hidx] = (__bf16)hnew;
    float* hs = dir ? hs_b : hs_f;
    hs[((size_t)b * Tt + tt) * Hh + j] = hnew;
}

__global__ __launch_bounds__(256)
void mean_feats(const float* __restrict__ hs_f, const float* __restrict__ hs_b,
                const float* __restrict__ mask, __bf16* __restrict__ F16) {
    const int Tt = 128, Hh = 512;
    const int idx = blockIdx.x * 256 + threadIdx.x;    // B*1024
    const int b = idx >> 10, d = idx & 1023;
    const float* hs = (d < Hh) ? hs_f : hs_b;
    const int dd = d & 511;
    float s = 0.0f, cnt = 0.0f;
    for (int t = 0; t < Tt; ++t) {
        const float m = mask[b * Tt + t];
        s   += hs[((size_t)b * Tt + t) * Hh + dd] * m;
        cnt += m;
    }
    F16[(size_t)b * 5120 + d] = (__bf16)(s / cnt);
}

__global__ __launch_bounds__(256)
void pack_conv_in(const float* __restrict__ hs_f, const float* __restrict__ hs_b,
                  const float* __restrict__ mask, __bf16* __restrict__ X) {
    const size_t idx = (size_t)blockIdx.x * 256 + threadIdx.x;  // B*T*1024
    const int d = (int)(idx & 1023);
    const size_t bt = idx >> 10;
    const float m = mask[bt];
    const float* hs = (d < 512) ? hs_f : hs_b;
    X[idx] = (__bf16)(hs[bt * 512 + (d & 511)] * m);
}

__global__ __launch_bounds__(256)
void bn_l2(float* __restrict__ out, const float* __restrict__ gamma,
           const float* __restrict__ beta, const float* __restrict__ mean,
           const float* __restrict__ var) {
    __shared__ float red[256];
    const int b = blockIdx.x;
    float ss = 0.0f;
    for (int j = threadIdx.x; j < 2048; j += 256) {
        const float x = out[(size_t)b * 2048 + j];
        const float v = gamma[j] * (x - mean[j]) * rsqrtf(var[j] + 1e-5f) + beta[j];
        out[(size_t)b * 2048 + j] = v;
        ss += v * v;
    }
    red[threadIdx.x] = ss;
    __syncthreads();
    for (int s = 128; s > 0; s >>= 1) {
        if (threadIdx.x < s) red[threadIdx.x] += red[threadIdx.x + s];
        __syncthreads();
    }
    const float inv = rsqrtf(red[0]);
    for (int j = threadIdx.x; j < 2048; j += 256)
        out[(size_t)b * 2048 + j] *= inv;
}

__global__ __launch_bounds__(256)
void cvt2d(const float* __restrict__ src, __bf16* __restrict__ dst,
           int R, int C, int ldD, int off) {
    const long long idx = (long long)blockIdx.x * 256 + threadIdx.x;
    if (idx >= (long long)R * C) return;
    const int r = (int)(idx / C), c = (int)(idx % C);
    dst[(size_t)r * ldD + off + c] = (__bf16)src[idx];
}

__global__ __launch_bounds__(256)
void cvt_t(const float* __restrict__ src, __bf16* __restrict__ dst, int R, int C) {
    const long long idx = (long long)blockIdx.x * 256 + threadIdx.x;
    if (idx >= (long long)R * C) return;
    const int r = (int)(idx / C), c = (int)(idx % C);
    dst[(size_t)c * R + r] = (__bf16)src[idx];
}

__global__ __launch_bounds__(256)
void zero_f32(float* __restrict__ p, long long n) {
    const long long idx = (long long)blockIdx.x * 256 + threadIdx.x;
    if (idx < n) p[idx] = 0.0f;
}

// ---------------------------------------------------------------------------

static inline unsigned blocks_for(long long n) {
    return (unsigned)((n + 255) / 256);
}

extern "C" void kernel_launch(void* const* d_in, const int* in_sizes, int n_in,
                              void* d_out, int out_size, void* d_ws, size_t ws_size,
                              hipStream_t stream) {
    (void)in_sizes; (void)n_in; (void)out_size; (void)ws_size;
    const int Bn = 128, Tt = 128, Dd = 2048, Hh = 512, Kc = 512;
    const int kw[4] = {2, 3, 4, 5};

    const float* videos   = (const float*)d_in[0];
    const float* vorigin  = (const float*)d_in[1];
    /* d_in[2] = lengths (recomputed from mask instead) */
    const float* mask     = (const float*)d_in[3];
    const float* w_ih_f   = (const float*)d_in[4];
    const float* w_hh_f   = (const float*)d_in[5];
    const float* b_ih_f   = (const float*)d_in[6];
    const float* b_hh_f   = (const float*)d_in[7];
    const float* w_ih_b   = (const float*)d_in[8];
    const float* w_hh_b   = (const float*)d_in[9];
    const float* b_ih_b   = (const float*)d_in[10];
    const float* b_hh_b   = (const float*)d_in[11];
    const float* conv_w[4] = {(const float*)d_in[12], (const float*)d_in[14],
                              (const float*)d_in[16], (const float*)d_in[18]};
    const float* conv_b[4] = {(const float*)d_in[13], (const float*)d_in[15],
                              (const float*)d_in[17], (const float*)d_in[19]};
    const float* fc_w     = (const float*)d_in[20];
    const float* fc_b     = (const float*)d_in[21];
    const float* bn_gamma = (const float*)d_in[22];
    const float* bn_beta  = (const float*)d_in[23];
    const float* bn_mean  = (const float*)d_in[24];
    const float* bn_var   = (const float*)d_in[25];
    float* out = (float*)d_out;

    char* wsp = (char*)d_ws;
    auto alloc = [&](size_t bytes) -> void* {
        void* p = wsp;
        wsp += (bytes + 255) & ~(size_t)255;
        return p;
    };

    __bf16* X16    = (__bf16*)alloc((size_t)Bn * Tt * Dd * 2);
    __bf16* WihT_f = (__bf16*)alloc((size_t)Dd * 3 * Hh * 2);
    __bf16* WihT_b = (__bf16*)alloc((size_t)Dd * 3 * Hh * 2);
    __bf16* WhhT   = (__bf16*)alloc((size_t)2 * Hh * 3 * Hh * 2);    // [2][H][3H]
    float*  xw_f   = (float*) alloc((size_t)Bn * Tt * 3 * Hh * 4);
    float*  xw_b   = (float*) alloc((size_t)Bn * Tt * 3 * Hh * 4);
    float*  gh     = (float*) alloc((size_t)2 * Bn * 3 * Hh * 4);
    float*  hst    = (float*) alloc((size_t)2 * Bn * Hh * 4);
    __bf16* hbf    = (__bf16*)alloc((size_t)2 * Bn * Hh * 2);
    float*  hs_f   = (float*) alloc((size_t)Bn * Tt * Hh * 4);
    float*  hs_b   = (float*) alloc((size_t)Bn * Tt * Hh * 4);
    __bf16* M16    = (__bf16*)alloc((size_t)Bn * Tt * 2 * Hh * 2);
    __bf16* CW16   = (__bf16*)alloc((size_t)14 * 1024 * Kc * 2);
    float*  conf   = (float*) alloc((size_t)Bn * 4 * Kc * 4);
    __bf16* F16    = (__bf16*)alloc((size_t)Bn * 5120 * 2);
    __bf16* FCW16  = (__bf16*)alloc((size_t)5120 * 2048 * 2);

    // ---- init + bf16 conversions -----------------------------------------
    zero_f32<<<blocks_for(2LL * Bn * Hh), 256, 0, stream>>>(hst, 2LL * Bn * Hh);
    zero_f32<<<blocks_for((2LL * Bn * Hh) / 2), 256, 0, stream>>>((float*)hbf, (2LL * Bn * Hh) / 2);
    zero_f32<<<blocks_for((long long)Bn * 4 * Kc), 256, 0, stream>>>(conf, (long long)Bn * 4 * Kc);

    cvt2d<<<blocks_for((long long)Bn * Tt * Dd), 256, 0, stream>>>(videos, X16, Bn * Tt, Dd, Dd, 0);
    cvt_t<<<blocks_for(3LL * Hh * Dd), 256, 0, stream>>>(w_ih_f, WihT_f, 3 * Hh, Dd);
    cvt_t<<<blocks_for(3LL * Hh * Dd), 256, 0, stream>>>(w_ih_b, WihT_b, 3 * Hh, Dd);
    cvt_t<<<blocks_for(3LL * Hh * Hh), 256, 0, stream>>>(w_hh_f, WhhT, 3 * Hh, Hh);
    cvt_t<<<blocks_for(3LL * Hh * Hh), 256, 0, stream>>>(w_hh_b, WhhT + (size_t)Hh * 3 * Hh, 3 * Hh, Hh);
    {
        size_t off = 0;
        for (int i = 0; i < 4; ++i) {
            const int wc = kw[i] * 1024;
            cvt_t<<<blocks_for((long long)Kc * wc), 256, 0, stream>>>(conv_w[i], CW16 + off, Kc, wc);
            off += (size_t)wc * Kc;
        }
    }
    cvt2d<<<blocks_for(5120LL * 2048), 256, 0, stream>>>(fc_w, FCW16, 5120, 2048, 2048, 0);

    // ---- x-projection GEMMs: [16384,2048]x[2048,1536] + b_ih --------------
    gemm_wmma_bf16<<<dim3(12, 256, 1), 256, 0, stream>>>(
        X16, WihT_f, b_ih_f, xw_f, Bn * Tt, 3 * Hh, Dd, 0, 0, 0, 0);
    gemm_wmma_bf16<<<dim3(12, 256, 1), 256, 0, stream>>>(
        X16, WihT_b, b_ih_b, xw_b, Bn * Tt, 3 * Hh, Dd, 0, 0, 0, 0);

    // ---- bidirectional GRU scan (both directions per step via grid.z) -----
    for (int t = 0; t < Tt; ++t) {
        gemm_wmma_bf16<<<dim3(12, 2, 2), 256, 0, stream>>>(
            hbf, WhhT, nullptr, gh, Bn, 3 * Hh, Hh,
            (long long)Bn * Hh, (long long)Hh * 3 * Hh, 0, (long long)Bn * 3 * Hh);
        gru_gate<<<512, 256, 0, stream>>>(xw_f, xw_b, gh, b_hh_f, b_hh_b,
                                          hst, hbf, hs_f, hs_b, t);
    }

    // ---- masked mean + conv input packing ---------------------------------
    mean_feats<<<blocks_for((long long)Bn * 1024), 256, 0, stream>>>(hs_f, hs_b, mask, F16);
    pack_conv_in<<<blocks_for((long long)Bn * Tt * 1024), 256, 0, stream>>>(hs_f, hs_b, mask, M16);

    // ---- convolutions: shifted-A WMMA GEMM + fused ReLU/max-pool ----------
    {
        size_t off = 0;
        for (int i = 0; i < 4; ++i) {
            conv_wmma<<<dim3(4, 3, Bn), 256, 0, stream>>>(
                M16, CW16 + off, conv_b[i], conf, kw[i], i * Kc);
            off += (size_t)kw[i] * 1024 * Kc;
        }
    }

    // ---- assemble feature vector (bf16) -----------------------------------
    cvt2d<<<blocks_for((long long)Bn * 2048), 256, 0, stream>>>(conf, F16, Bn, 2048, 5120, 1024);
    cvt2d<<<blocks_for((long long)Bn * 2048), 256, 0, stream>>>(vorigin, F16, Bn, 2048, 5120, 3072);

    // ---- FC: [128,5120]x[5120,2048] + fc_b -> d_out -----------------------
    gemm_wmma_bf16<<<dim3(16, 2, 1), 256, 0, stream>>>(
        F16, FCW16, fc_b, out, Bn, 2048, 5120, 0, 0, 0, 0);

    // ---- BatchNorm + L2 normalize (in place) ------------------------------
    bn_l2<<<Bn, 256, 0, stream>>>(out, bn_gamma, bn_beta, bn_mean, bn_var);
}
